// GNN_41128606826949
// MI455X (gfx1250) — matrix operand order repaired and checked by reading, not compiled, not run on hardware
//
#include <hip/hip_runtime.h>
#include <hip/hip_bf16.h>
#include <stdint.h>

#define THREADS 256

// --- CDNA5 WMMA types -------------------------------------------------------
typedef __attribute__((ext_vector_type(16))) __bf16       v16bf;
typedef __attribute__((ext_vector_type(8)))  float        v8f;
typedef __attribute__((ext_vector_type(4)))  unsigned int u32x4;

union Frag256 {                 // 32 bytes: 16 bf16 elements == 8 VGPRs
  struct { u32x4 lo, hi; } s;
  v16bf v;
};

__device__ __forceinline__ unsigned short f2bf(float f) {
  unsigned int u = __float_as_uint(f);
  u += 0x7FFFu + ((u >> 16) & 1u);   // round-to-nearest-even
  return (unsigned short)(u >> 16);
}

// --- graph normalization ----------------------------------------------------
__global__ void k_deg_init(float* deg, int n) {
  int i = blockIdx.x * blockDim.x + threadIdx.x;
  if (i < n) deg[i] = 1.0f;                       // self-loop contribution
}
__global__ void k_deg_edges(const int* __restrict__ dst, float* deg, int e) {
  int i = blockIdx.x * blockDim.x + threadIdx.x;
  if (i < e) atomicAdd(&deg[dst[i]], 1.0f);
}
__global__ void k_dinv(float* d, int n) {
  int i = blockIdx.x * blockDim.x + threadIdx.x;
  if (i < n) { float v = d[i]; d[i] = v > 0.f ? rsqrtf(v) : 0.f; }
}

// --- fp32 -> bf16 feature conversion ---------------------------------------
__global__ void k_f32_to_bf16(const float* __restrict__ x,
                              unsigned short* __restrict__ o, long long n4) {
  long long t = (long long)blockIdx.x * blockDim.x + threadIdx.x;
  if (t >= n4) return;
  float4 v = ((const float4*)x)[t];
  ushort4 r;
  r.x = f2bf(v.x); r.y = f2bf(v.y); r.z = f2bf(v.z); r.w = f2bf(v.w);
  ((ushort4*)o)[t] = r;
}

// --- bf16 WMMA GEMM:  C[M][N] = A[M][128] * W[128][N]  ----------------------
// N is compile-time so the accumulator tile array fully unrolls into VGPRs
// (runtime N caused the compiler to demote acc[] to scratch).
// 8 waves/block, each wave computes a 16 x N strip; W staged transposed in LDS.
template <int N>
__global__ __launch_bounds__(THREADS) void k_gemm_bf16(
    const unsigned short* __restrict__ A,   // [M][128] bf16
    const float* __restrict__ W,            // [128][N] f32 row-major ([in,out])
    float* __restrict__ C,                  // [M][N]  f32
    int M) {
  constexpr int K    = 128;
  constexpr int LSTR = 136;                 // padded stride: conflict-free ds_load_b128
  constexpr int NT   = N >> 4;              // 8 (N=128) or 4 (N=64) accumulator tiles
  __shared__ unsigned short Wt[128 * LSTR]; // Wt[n][k] = bf16(W[k][n]), 34.8 KB

  const int tid = threadIdx.x;
  for (int i = tid; i < K * N; i += THREADS) {  // coalesced W read
    int k = i / N, n = i % N;
    Wt[n * LSTR + k] = f2bf(W[i]);
  }
  __syncthreads();

  const int wave = tid >> 5;
  const int lane = tid & 31;
  const int m    = lane & 15;
  const int hi   = lane >> 4;               // lane half selects K sub-range
  const int row0 = blockIdx.x * 128 + wave * 16;
  const int rowA = min(row0 + m, M - 1);    // clamp: EXEC stays all-ones for WMMA
  const unsigned short* arow = A + (size_t)rowA * K;

  v8f zero = {};
  v8f acc[NT];
#pragma unroll
  for (int nt = 0; nt < NT; ++nt) acc[nt] = zero;

#pragma unroll
  for (int k0 = 0; k0 < K; k0 += 32) {
    // A frag (16x32 bf16): lanes 0-15 hold K+0..7 / K+16..23, lanes 16-31 K+8..15 / K+24..31
    Frag256 a;
    a.s.lo = *(const u32x4*)(arow + k0 + (hi ? 8  : 0));
    a.s.hi = *(const u32x4*)(arow + k0 + (hi ? 24 : 16));
#pragma unroll
    for (int nt = 0; nt < NT; ++nt) {
      // B frag (32x16 bf16): col = lane%16; lanes 0-15 K+0..7 / K+16..23, lanes 16-31 K+8..15 / K+24..31
      const unsigned short* wp = Wt + (size_t)(nt * 16 + m) * LSTR + k0 + (hi ? 8 : 0);
      Frag256 b;
      b.s.lo = *(const u32x4*)(wp);
      b.s.hi = *(const u32x4*)(wp + 16);
      acc[nt] = __builtin_amdgcn_wmma_f32_16x16x32_bf16(
          false, a.v, false, b.v, (short)0, acc[nt], false, false);
    }
  }

  // D layout: VGPR r, lanes 0-15 -> M=r, lanes 16-31 -> M=r+8; col = lane%16
#pragma unroll
  for (int nt = 0; nt < NT; ++nt) {
    int col = nt * 16 + m;
#pragma unroll
    for (int r = 0; r < 8; ++r) {
      int rr = row0 + r + (hi ? 8 : 0);
      if (rr < M) C[(size_t)rr * N + col] = acc[nt][r];
    }
  }
}

// --- agg init with self-loop term: agg = dinv[i]^2 * h ----------------------
__global__ void k_self_agg(const float* __restrict__ h, const float* __restrict__ dinv,
                           float* __restrict__ agg, long long tot4, int cpe) {
  long long t = (long long)blockIdx.x * blockDim.x + threadIdx.x;
  if (t >= tot4) return;
  int row = (int)(t / cpe);
  float d = dinv[row], w = d * d;
  float4 v = ((const float4*)h)[t];
  v.x *= w; v.y *= w; v.z *= w; v.w *= w;
  ((float4*)agg)[t] = v;
}

// --- edge gather/scale/scatter: one wave per edge (F=128) -------------------
__global__ void k_scatter(const float* __restrict__ h, const float* __restrict__ dinv,
                          const int* __restrict__ src, const int* __restrict__ dst,
                          float* __restrict__ agg, long long E, int F) {
  long long t = (long long)blockIdx.x * blockDim.x + threadIdx.x;
  int cpe = F >> 2;
  if (t >= E * cpe) return;
  long long e = t / cpe;
  int c = (int)(t - e * cpe);
  int s = src[e], d = dst[e];
  float w = dinv[s] * dinv[d];
  float4 v = ((const float4*)(h + (size_t)s * F))[c];
  float* o = agg + (size_t)d * F + (size_t)c * 4;
  atomicAdd(o + 0, w * v.x);
  atomicAdd(o + 1, w * v.y);
  atomicAdd(o + 2, w * v.z);
  atomicAdd(o + 3, w * v.w);
}

// --- bias + relu + dropout(p=0.5) -> bf16 next-layer features ---------------
__global__ void k_post(const float* __restrict__ agg, const float* __restrict__ b,
                       unsigned short* __restrict__ obf, long long n, int F,
                       unsigned seed) {
  long long t = (long long)blockIdx.x * blockDim.x + threadIdx.x;
  if (t >= n) return;
  int col = (int)(t % F);
  float v = agg[t] + b[col];
  v = v > 0.f ? v : 0.f;
  unsigned hsh = (unsigned)t * 2654435761u + seed * 0x9E3779B9u;
  hsh ^= hsh >> 16; hsh *= 2246822519u; hsh ^= hsh >> 13;
  v = (hsh & 1u) ? v * 2.0f : 0.0f;
  obf[t] = f2bf(v);
}

// --- final layer: out = agg + b3 (fp32) -------------------------------------
__global__ void k_final(const float* __restrict__ agg, const float* __restrict__ b,
                        float* __restrict__ out, long long n, int F) {
  long long t = (long long)blockIdx.x * blockDim.x + threadIdx.x;
  if (t >= n) return;
  out[t] = agg[t] + b[(int)(t % F)];
}

extern "C" void kernel_launch(void* const* d_in, const int* in_sizes, int n_in,
                              void* d_out, int out_size, void* d_ws, size_t ws_size,
                              hipStream_t stream) {
  const float* x  = (const float*)d_in[0];
  const float* W1 = (const float*)d_in[1];
  const float* b1 = (const float*)d_in[2];
  const float* W2 = (const float*)d_in[3];
  const float* b2 = (const float*)d_in[4];
  const float* W3 = (const float*)d_in[5];
  const float* b3 = (const float*)d_in[6];
  const int*   ei = (const int*)d_in[7];

  const int IN = 128, HID = 128, OUT = 64;
  const int Nn = in_sizes[0] / IN;
  const long long E = in_sizes[7] / 2;
  const int* src = ei;
  const int* dst = ei + E;

  char* ws = (char*)d_ws;
  size_t off = 0;
  auto alloc = [&](size_t bytes) -> void* {
    void* p = ws + off;
    off += (bytes + 255) & ~(size_t)255;
    return p;
  };
  float*          dinv   = (float*)alloc((size_t)Nn * 4);
  unsigned short* featbf = (unsigned short*)alloc((size_t)Nn * HID * 2);
  float*          gem    = (float*)alloc((size_t)Nn * HID * 4);
  float*          agg    = (float*)alloc((size_t)Nn * HID * 4);

  auto cdiv = [](long long a, long long b) -> unsigned { return (unsigned)((a + b - 1) / b); };

  // degree -> dinv (self-loops folded in as init 1.0)
  k_deg_init <<<cdiv(Nn, 256), 256, 0, stream>>>(dinv, Nn);
  k_deg_edges<<<cdiv(E, 256),  256, 0, stream>>>(dst, dinv, (int)E);
  k_dinv     <<<cdiv(Nn, 256), 256, 0, stream>>>(dinv, Nn);

  // x -> bf16 features
  long long n4 = (long long)Nn * IN / 4;
  k_f32_to_bf16<<<cdiv(n4, 256), 256, 0, stream>>>(x, featbf, n4);

  const unsigned gblk = (unsigned)((Nn + 127) / 128);

  // ---- layer 1 (128 -> 128) ----
  k_gemm_bf16<128><<<gblk, THREADS, 0, stream>>>(featbf, W1, gem, Nn);
  k_self_agg <<<cdiv((long long)Nn * HID / 4, 256), 256, 0, stream>>>(gem, dinv, agg,
                                                     (long long)Nn * HID / 4, HID / 4);
  k_scatter  <<<cdiv(E * (HID / 4), 256), 256, 0, stream>>>(gem, dinv, src, dst, agg, E, HID);
  k_post     <<<cdiv((long long)Nn * HID, 256), 256, 0, stream>>>(agg, b1, featbf,
                                                     (long long)Nn * HID, HID, 1u);
  // ---- layer 2 (128 -> 128) ----
  k_gemm_bf16<128><<<gblk, THREADS, 0, stream>>>(featbf, W2, gem, Nn);
  k_self_agg <<<cdiv((long long)Nn * HID / 4, 256), 256, 0, stream>>>(gem, dinv, agg,
                                                     (long long)Nn * HID / 4, HID / 4);
  k_scatter  <<<cdiv(E * (HID / 4), 256), 256, 0, stream>>>(gem, dinv, src, dst, agg, E, HID);
  k_post     <<<cdiv((long long)Nn * HID, 256), 256, 0, stream>>>(agg, b2, featbf,
                                                     (long long)Nn * HID, HID, 2u);
  // ---- layer 3 (128 -> 64) ----
  k_gemm_bf16<64><<<gblk, THREADS, 0, stream>>>(featbf, W3, gem, Nn);
  k_self_agg <<<cdiv((long long)Nn * OUT / 4, 256), 256, 0, stream>>>(gem, dinv, agg,
                                                     (long long)Nn * OUT / 4, OUT / 4);
  k_scatter  <<<cdiv(E * (OUT / 4), 256), 256, 0, stream>>>(gem, dinv, src, dst, agg, E, OUT);
  k_final    <<<cdiv((long long)Nn * OUT, 256), 256, 0, stream>>>(agg, b3, (float*)d_out,
                                                     (long long)Nn * OUT, OUT);
}
